// Int8BloomAttention_7421703487912
// MI455X (gfx1250) — 8-bit matrix operand order repaired and checked by reading, compiled, not run on hardware
//
#include <hip/hip_runtime.h>
#include <hip/hip_bf16.h>
#include <cstdint>
#include <cstddef>


typedef int v8i __attribute__((ext_vector_type(8)));

#define QKV_ALPHA   1e-4f
#define QK_SCALE    1e-4f
#define PV_SCALE    2e-3f
#define DENSE_ALPHA 1e-5f
#define INV_NORM    0.08838834764831845f
#define FMIN_F32    (-3.4028234663852886e38f)

#define WMMA_IU8(a, b, c) \
  __builtin_amdgcn_wmma_i32_16x16x64_iu8(true, (a), true, (b), (c), false, false)

static __device__ inline v8i v8i_zero() {
  v8i z;
#pragma unroll
  for (int i = 0; i < 8; ++i) z[i] = 0;
  return z;
}

static __device__ inline v8i load_a_frag(const unsigned char* tile, int ldk,
                                         int lane, int k0) {
  int row = lane & 15;
  int kb  = (lane & 16) ? 8 : 0;
  const unsigned char* p = tile + row * ldk + k0 + kb;
  v8i a;
#pragma unroll
  for (int v = 0; v < 8; ++v) {
    int off = ((v >> 1) << 4) + ((v & 1) << 2);
    a[v] = *(const int*)(p + off);
  }
  return a;
}

static __device__ inline v8i load_b_frag(const unsigned char* tile, int ldk,
                                         int lane, int col0, int k0) {
  int col = lane & 15;
  int kb  = (lane & 16) ? 8 : 0;
  const unsigned char* p = tile + (col0 + col) * ldk + k0 + kb;
  v8i b;
#pragma unroll
  for (int v = 0; v < 8; ++v) {
    int off = ((v >> 1) << 4) + ((v & 1) << 2);
    b[v] = *(const int*)(p + off);
  }
  return b;
}

static __device__ inline int quant_i8(float x) {
  float r = rintf(x);
  r = fminf(fmaxf(r, -128.0f), 127.0f);
  return (int)r;
}

__global__ void pack_i8_kernel(const int* __restrict__ src,
                               signed char* __restrict__ dst, int n) {
  int i = blockIdx.x * blockDim.x + threadIdx.x;
  if (i < n) dst[i] = (signed char)src[i];
}

__global__ __launch_bounds__(128)
__attribute__((amdgpu_waves_per_eu(1))) void qkv_gemm_kernel(
    const signed char* __restrict__ A8,
    const signed char* __restrict__ W8,
    const signed char* __restrict__ bias8,
    signed char* __restrict__ qkvout) {
  constexpr int K = 2048, TM = 64, TN = 128, TK = 64, NIT = K / TK;
  constexpr int NT = 6144 / TN;
  constexpr int ABUF = TM * TK, BBUF = TN * TK;
  const int mt = blockIdx.x / NT, nt = blockIdx.x % NT;
  const int m0 = mt * TM, n0 = nt * TN;
  const int t = threadIdx.x, lane = t & 31, wave = t >> 5;

  __shared__ __align__(16) unsigned char At[2][TM][TK];
  __shared__ __align__(16) unsigned char Bt[2][TN][TK];

  const signed char* gA = A8 + (size_t)(m0 + (t >> 1)) * K + (t & 1) * 32;
  const signed char* gB = W8 + (size_t)(n0 + t) * K;
  unsigned char* lA = &At[0][t >> 1][(t & 1) * 32];
  unsigned char* lB = &Bt[0][t][0];

  v8i acc[8];
#pragma unroll
  for (int i = 0; i < 8; ++i) acc[i] = v8i_zero();

  {
    uint4 a0 = *(const uint4*)(gA);
    uint4 a1 = *(const uint4*)(gA + 16);
    uint4 b0 = *(const uint4*)(gB);
    uint4 b1 = *(const uint4*)(gB + 16);
    uint4 b2 = *(const uint4*)(gB + 32);
    uint4 b3 = *(const uint4*)(gB + 48);
    *(uint4*)(lA) = a0;      *(uint4*)(lA + 16) = a1;
    *(uint4*)(lB) = b0;      *(uint4*)(lB + 16) = b1;
    *(uint4*)(lB + 32) = b2; *(uint4*)(lB + 48) = b3;
  }

  for (int it = 0; it < NIT; ++it) {
    __syncthreads();
    const int buf = it & 1, nbuf = buf ^ 1;
    const bool pf = (it + 1 < NIT);
    uint4 a0, a1, b0, b1, b2, b3;
    if (pf) {
      int k0 = (it + 1) * TK;
      a0 = *(const uint4*)(gA + k0);
      a1 = *(const uint4*)(gA + k0 + 16);
      b0 = *(const uint4*)(gB + k0);
      b1 = *(const uint4*)(gB + k0 + 16);
      b2 = *(const uint4*)(gB + k0 + 32);
      b3 = *(const uint4*)(gB + k0 + 48);
      if (it + 4 < NIT)
        __builtin_prefetch(gB + (it + 4) * TK, 0, 1);
    }

    v8i a = load_a_frag(&At[buf][wave * 16][0], TK, lane, 0);
    v8i bcur = load_b_frag(&Bt[buf][0][0], TK, lane, 0, 0);
#pragma unroll
    for (int ns = 0; ns < 8; ++ns) {
      v8i bnext;
      if (ns < 7) bnext = load_b_frag(&Bt[buf][0][0], TK, lane, (ns + 1) * 16, 0);
      acc[ns] = WMMA_IU8(a, bcur, acc[ns]);
      if (ns < 7) bcur = bnext;
    }

    if (pf) {
      *(uint4*)(lA + nbuf * ABUF) = a0;
      *(uint4*)(lA + nbuf * ABUF + 16) = a1;
      *(uint4*)(lB + nbuf * BBUF) = b0;
      *(uint4*)(lB + nbuf * BBUF + 16) = b1;
      *(uint4*)(lB + nbuf * BBUF + 32) = b2;
      *(uint4*)(lB + nbuf * BBUF + 48) = b3;
    }
  }

  const int col = lane & 15;
  const int mbase = m0 + wave * 16 + ((lane & 16) ? 8 : 0);
#pragma unroll
  for (int ns = 0; ns < 8; ++ns) {
    int o = n0 + ns * 16 + col;
    float bf = (float)bias8[o];
    int nh = o / 384, rem = o % 384;
    int which = rem / 128, hd = rem % 128;
#pragma unroll
    for (int v = 0; v < 8; ++v) {
      int m = mbase + v;
      int b = m >> 10, qq = m & 1023;
      signed char val = (signed char)quant_i8((float)acc[ns][v] * QKV_ALPHA + bf);
      int head = b * 16 + nh;
      if (which == 2)
        qkvout[(size_t)2 * 4194304 + ((size_t)(head * 128 + hd)) * 1024 + qq] = val;
      else
        qkvout[(size_t)which * 4194304 + ((size_t)(head * 1024 + qq)) * 128 + hd] = val;
    }
  }
}

__global__ __launch_bounds__(128)
__attribute__((amdgpu_waves_per_eu(1))) void attn_kernel(
    const signed char* __restrict__ q8, const signed char* __restrict__ k8,
    const signed char* __restrict__ vt8,
    const float* __restrict__ alibi, signed char* __restrict__ ctx8) {
  constexpr int Q = 1024, HD = 128;
  const int wg = blockIdx.x;
  const int n  = wg >> 6;
  const int qt0 = (wg & 63) * 16;
  const int t = threadIdx.x, lane = t & 31, wave = t >> 5;

  __shared__ __align__(16) float s[16][1024];
  __shared__ __align__(16) unsigned char p[16][1024];
  __shared__ __align__(16) unsigned char qtile[16][128];
  __shared__ __align__(16) unsigned char tiles[2][128][64];
  __shared__ float red[16][8];
  __shared__ float rowmax[16], rowsum[16];

  {
    int row = t >> 3, off = (t & 7) << 4;
    *(uint4*)&qtile[row][off] =
        *(const uint4*)(q8 + ((size_t)(n * Q + qt0 + row)) * HD + off);
  }
  __syncthreads();
  v8i a0 = load_a_frag(&qtile[0][0], 128, lane, 0);
  v8i a1 = load_a_frag(&qtile[0][0], 128, lane, 64);

  unsigned char* tflat = &tiles[0][0][0];
  for (int kt = wave; kt < 64; kt += 4) {
    unsigned char* ktile = tflat + wave * 2048;
    {
      int key = lane & 15, half = lane >> 4;
      const uint4* src =
          (const uint4*)(k8 + ((size_t)(n * Q + kt * 16 + key)) * HD + half * 64);
      uint4* dst = (uint4*)&ktile[key * 128 + half * 64];
      dst[0] = src[0]; dst[1] = src[1]; dst[2] = src[2]; dst[3] = src[3];
    }
    v8i b0 = load_b_frag(ktile, 128, lane, 0, 0);
    v8i b1 = load_b_frag(ktile, 128, lane, 0, 64);
    v8i c0 = v8i_zero(), c1 = v8i_zero();
    c0 = WMMA_IU8(a0, b0, c0);
    c1 = WMMA_IU8(a1, b1, c1);

    int col = lane & 15;
    int gk = kt * 16 + col;
    float al = alibi[(size_t)n * Q + gk];
#pragma unroll
    for (int v = 0; v < 8; ++v) {
      int row = v + ((lane & 16) ? 8 : 0);
      int gq = qt0 + row;
      float f = (gk > gq)
                    ? FMIN_F32
                    : (al + (float)(c0[v] + c1[v]) * (QK_SCALE * INV_NORM));
      s[row][gk] = f;
    }
  }
  __syncthreads();

  {
    int row = t >> 3, part = t & 7;
    int c0 = part * 128;
    float m = FMIN_F32;
    for (int c = 0; c < 128; ++c) m = fmaxf(m, s[row][c0 + c]);
    red[row][part] = m;
    __syncthreads();
    if (part == 0) {
      float mm = red[row][0];
      for (int i = 1; i < 8; ++i) mm = fmaxf(mm, red[row][i]);
      rowmax[row] = mm;
    }
    __syncthreads();
    float mm = rowmax[row];
    float sum = 0.0f;
    for (int c = 0; c < 128; ++c) {
      float e = __expf(s[row][c0 + c] - mm);
      s[row][c0 + c] = e;
      sum += e;
    }
    red[row][part] = sum;
    __syncthreads();
    if (part == 0) {
      float ss = 0.0f;
      for (int i = 0; i < 8; ++i) ss += red[row][i];
      rowsum[row] = ss;
    }
    __syncthreads();
    float inv = 1.0f / rowsum[row];
    for (int c = 0; c < 128; ++c) {
      int pi = quant_i8(s[row][c0 + c] * inv * 127.0f);
      p[row][c0 + c] = (unsigned char)pi;
    }
  }

  const uint4* gV = (const uint4*)(vt8 + ((size_t)(n * HD + t)) * 1024);
  uint4* lV0 = (uint4*)&tiles[0][t][0];
  uint4* lV1 = (uint4*)&tiles[1][t][0];

  v8i pacc[2];
  pacc[0] = v8i_zero();
  pacc[1] = v8i_zero();

  {
    uint4 v0 = gV[0], v1 = gV[1], v2 = gV[2], v3 = gV[3];
    lV0[0] = v0; lV0[1] = v1; lV0[2] = v2; lV0[3] = v3;
  }

  for (int kt2 = 0; kt2 < 16; ++kt2) {
    __syncthreads();
    const int buf = kt2 & 1;
    const bool pf = (kt2 + 1 < 16);
    uint4 v0, v1, v2, v3;
    if (pf) {
      const uint4* src = gV + (kt2 + 1) * 4;
      v0 = src[0]; v1 = src[1]; v2 = src[2]; v3 = src[3];
    }

    v8i pa = load_a_frag(&p[0][0], 1024, lane, kt2 * 64);
#pragma unroll
    for (int j = 0; j < 2; ++j) {
      v8i vb = load_b_frag(&tiles[buf][0][0], 64, lane, (wave * 2 + j) * 16, 0);
      pacc[j] = WMMA_IU8(pa, vb, pacc[j]);
    }

    if (pf) {
      uint4* dst = buf ? lV0 : lV1;
      dst[0] = v0; dst[1] = v1; dst[2] = v2; dst[3] = v3;
    }
  }

  {
    int col = lane & 15;
    int b = n >> 4, nh = n & 15;
#pragma unroll
    for (int j = 0; j < 2; ++j) {
      int d = (wave * 2 + j) * 16 + col;
#pragma unroll
      for (int v = 0; v < 8; ++v) {
        int row = v + ((lane & 16) ? 8 : 0);
        int gq = qt0 + row;
        int c8 = quant_i8((float)pacc[j][v] * PV_SCALE);
        ctx8[((size_t)(b * Q + gq)) * 2048 + nh * 128 + d] = (signed char)c8;
      }
    }
  }
}

__global__ __launch_bounds__(128)
__attribute__((amdgpu_waves_per_eu(1))) void dense_gemm_kernel(
    const signed char* __restrict__ A8,
    const signed char* __restrict__ W8,
    const float* __restrict__ bden,
    const float* __restrict__ resid,
    float* __restrict__ out) {
  constexpr int K = 2048, H = 2048, TM = 64, TN = 128, TK = 64, NIT = K / TK;
  constexpr int NT = H / TN;
  constexpr int ABUF = TM * TK, BBUF = TN * TK;
  const int mt = blockIdx.x / NT, nt = blockIdx.x % NT;
  const int m0 = mt * TM, n0 = nt * TN;
  const int t = threadIdx.x, lane = t & 31, wave = t >> 5;

  __shared__ __align__(16) unsigned char At[2][TM][TK];
  __shared__ __align__(16) unsigned char Bt[2][TN][TK];

  const signed char* gA = A8 + (size_t)(m0 + (t >> 1)) * K + (t & 1) * 32;
  const signed char* gB = W8 + (size_t)(n0 + t) * K;
  unsigned char* lA = &At[0][t >> 1][(t & 1) * 32];
  unsigned char* lB = &Bt[0][t][0];

  v8i acc[8];
#pragma unroll
  for (int i = 0; i < 8; ++i) acc[i] = v8i_zero();

  {
    uint4 a0 = *(const uint4*)(gA);
    uint4 a1 = *(const uint4*)(gA + 16);
    uint4 b0 = *(const uint4*)(gB);
    uint4 b1 = *(const uint4*)(gB + 16);
    uint4 b2 = *(const uint4*)(gB + 32);
    uint4 b3 = *(const uint4*)(gB + 48);
    *(uint4*)(lA) = a0;      *(uint4*)(lA + 16) = a1;
    *(uint4*)(lB) = b0;      *(uint4*)(lB + 16) = b1;
    *(uint4*)(lB + 32) = b2; *(uint4*)(lB + 48) = b3;
  }

  for (int it = 0; it < NIT; ++it) {
    __syncthreads();
    const int buf = it & 1, nbuf = buf ^ 1;
    const bool pf = (it + 1 < NIT);
    uint4 a0, a1, b0, b1, b2, b3;
    if (pf) {
      int k0 = (it + 1) * TK;
      a0 = *(const uint4*)(gA + k0);
      a1 = *(const uint4*)(gA + k0 + 16);
      b0 = *(const uint4*)(gB + k0);
      b1 = *(const uint4*)(gB + k0 + 16);
      b2 = *(const uint4*)(gB + k0 + 32);
      b3 = *(const uint4*)(gB + k0 + 48);
      if (it + 4 < NIT) __builtin_prefetch(gB + (it + 4) * TK, 0, 1);
    }

    v8i a = load_a_frag(&At[buf][wave * 16][0], TK, lane, 0);
    v8i bcur = load_b_frag(&Bt[buf][0][0], TK, lane, 0, 0);
#pragma unroll
    for (int ns = 0; ns < 8; ++ns) {
      v8i bnext;
      if (ns < 7) bnext = load_b_frag(&Bt[buf][0][0], TK, lane, (ns + 1) * 16, 0);
      acc[ns] = WMMA_IU8(a, bcur, acc[ns]);
      if (ns < 7) bcur = bnext;
    }

    if (pf) {
      *(uint4*)(lA + nbuf * ABUF) = a0;
      *(uint4*)(lA + nbuf * ABUF + 16) = a1;
      *(uint4*)(lB + nbuf * BBUF) = b0;
      *(uint4*)(lB + nbuf * BBUF + 16) = b1;
      *(uint4*)(lB + nbuf * BBUF + 32) = b2;
      *(uint4*)(lB + nbuf * BBUF + 48) = b3;
    }
  }

  const int col = lane & 15;
  const int mbase = m0 + wave * 16 + ((lane & 16) ? 8 : 0);
#pragma unroll
  for (int ns = 0; ns < 8; ++ns) {
    int o = n0 + ns * 16 + col;
    float bf = bden[o];
#pragma unroll
    for (int v = 0; v < 8; ++v) {
      int m = mbase + v;
      size_t idx = (size_t)m * H + o;
      out[idx] = (float)acc[ns][v] * DENSE_ALPHA + bf + resid[idx];
    }
  }
}

extern "C" void kernel_launch(void* const* d_in, const int* in_sizes, int n_in,
                              void* d_out, int out_size, void* d_ws,
                              size_t ws_size, hipStream_t stream) {
  const int*   hs_i32   = (const int*)d_in[0];
  const float* residual = (const float*)d_in[1];
  const float* alibi    = (const float*)d_in[2];
  const int*   wqkv_i32 = (const int*)d_in[4];
  const int*   bqkv_i32 = (const int*)d_in[5];
  const int*   wd_i32   = (const int*)d_in[6];
  const float* b_dense  = (const float*)d_in[7];
  float* out = (float*)d_out;

  unsigned char* ws = (unsigned char*)d_ws;
  signed char* hs8   = (signed char*)(ws);
  signed char* wqkv8 = (signed char*)(ws + ((size_t)4 << 20));
  signed char* wd8   = (signed char*)(ws + ((size_t)16 << 20));
  signed char* bqkv8 = (signed char*)(ws + ((size_t)20 << 20));
  signed char* q8    = (signed char*)(ws + ((size_t)21 << 20));
  signed char* k8    = q8 + 32 * 1024 * 128;
  signed char* v8t   = k8 + 32 * 1024 * 128;
  signed char* ctx8  = (signed char*)(ws + ((size_t)33 << 20));

  const int nHS = 2 * 1024 * 2048;
  const int nWQ = 6144 * 2048;
  const int nBQ = 6144;
  const int nWD = 2048 * 2048;

  pack_i8_kernel<<<(nHS + 255) / 256, 256, 0, stream>>>(hs_i32, hs8, nHS);
  pack_i8_kernel<<<(nWQ + 255) / 256, 256, 0, stream>>>(wqkv_i32, wqkv8, nWQ);
  pack_i8_kernel<<<(nBQ + 255) / 256, 256, 0, stream>>>(bqkv_i32, bqkv8, nBQ);
  pack_i8_kernel<<<(nWD + 255) / 256, 256, 0, stream>>>(wd_i32, wd8, nWD);

  qkv_gemm_kernel<<<32 * 48, 128, 0, stream>>>(hs8, wqkv8, bqkv8, q8);

  attn_kernel<<<32 * 64, 128, 0, stream>>>(q8, k8, v8t, alibi, ctx8);

  dense_gemm_kernel<<<32 * 16, 128, 0, stream>>>(ctx8, wd8, b_dense, residual,
                                                 out);
}
